// CrossViewSwapAttention_33277406609432
// MI455X (gfx1250) — matrix operand order repaired and checked by reading, not compiled, not run on hardware
//
#include <hip/hip_runtime.h>

typedef __attribute__((ext_vector_type(16))) _Float16 v16h;
typedef __attribute__((ext_vector_type(8)))  _Float16 v8h;
typedef __attribute__((ext_vector_type(8)))  float    v8f;
typedef __attribute__((ext_vector_type(4)))  unsigned int u32x4;
typedef __attribute__((ext_vector_type(4)))  int i32x4;
typedef __attribute__((ext_vector_type(8)))  int i32x8;

// ---------------------------------------------------------------------------
// Problem constants (from reference)
// ---------------------------------------------------------------------------
#define BB    4
#define NCAM  6
#define DIMC  128
#define NHEAD 4
#define DHEAD 32
#define LWIN  16          // windows per image (4x4)
#define MQ    1536        // merged queries per window  (6*16*16)
#define MK    96          // merged keys per window     (6*4*4)
#define TQ    ((long)BB*LWIN*MQ)     // 98304 q tokens
#define TKV   ((long)BB*LWIN*MK)     // 6144  kv tokens
#define THW   ((long)BB*64*64)       // 16384 spatial tokens
#define TF    (BB*NCAM*16*16)        // 6144 feature tokens
#define EPSF  1e-5f

// ---------------------------------------------------------------------------
// WMMA helpers (gfx1250 16x16x32 f16 -> f32). Layouts per CDNA5 ISA 7.12.2.
// ---------------------------------------------------------------------------
__device__ inline v16h wmma_load_a16(const _Float16* base, int ld) {
  int lane = threadIdx.x & 31;
  int m = lane & 15, hf2 = lane >> 4;
  const _Float16* p = base + (long)m * ld + hf2 * 8;
  v8h lo = *(const v8h*)p;
  v8h hi = *(const v8h*)(p + 16);
  v16h r;
#pragma unroll
  for (int i = 0; i < 8; ++i) { r[i] = lo[i]; r[i + 8] = hi[i]; }
  return r;
}

// B from column-major storage: bcm[n*ld + k]  (k contiguous, two 16B loads, same k order as A)
__device__ inline v16h wmma_load_b_cm(const _Float16* bcm, int ld) {
  int lane = threadIdx.x & 31;
  int n = lane & 15, hf2 = lane >> 4;
  const _Float16* p = bcm + (long)n * ld + hf2 * 8;
  v8h lo = *(const v8h*)p;
  v8h hi = *(const v8h*)(p + 16);
  v16h r;
#pragma unroll
  for (int i = 0; i < 8; ++i) { r[i] = lo[i]; r[i + 8] = hi[i]; }
  return r;
}

__device__ inline float wsum32(float v) {
#pragma unroll
  for (int o = 16; o > 0; o >>= 1) v += __shfl_xor(v, o, 32);
  return v;
}

__device__ inline float gelu_exact(float x) {
  return 0.5f * x * (1.0f + erff(x * 0.70710678118654752f));
}

// ---------------------------------------------------------------------------
// Tensor Data Mover: DMA a contiguous byte range (n8 x 8B elements) from
// global memory into LDS at byte offset lds_off. Minimal 1-D D# per CDNA5 ISA
// ch.8: group0 = {count=1, lds_addr, global_addr, type=2}; group1 carries
// data_size=8B, tensor_dim0 = tile_dim0 = n8 (so nothing is OOB-clipped),
// tensor_dim1 = 1. Groups 2/3 unused (<=2D tensor). Tracked by TENSORcnt.
// This toolchain's builtin takes 6 args (g0, g1, g2, g3, g4, cpol).
// ---------------------------------------------------------------------------
__device__ inline void tdm_load_tile(const void* gsrc, unsigned lds_off, unsigned n8) {
  unsigned long long ga = (unsigned long long)gsrc;
  u32x4 g0;
  g0[0] = 1u;                                   // [1:0] count=1 (valid user D#)
  g0[1] = lds_off;                              // [63:32] lds_addr
  g0[2] = (unsigned)ga;                         // [95:64] global_addr lo
  g0[3] = (unsigned)(ga >> 32) | (2u << 30);    // [120:96] addr hi | type=2
  i32x8 g1;
  g1[0] = 3 << 16;                              // data_size = 8 bytes
  g1[1] = (int)((n8 & 0xFFFFu) << 16);          // tensor_dim0[15:0]  @ bit 48
  g1[2] = (int)(((n8 >> 16) & 0xFFFFu) | (1u << 16)); // tensor_dim0 hi, tensor_dim1=1
  g1[3] = (int)((n8 & 0xFFFFu) << 16);          // tile_dim0 @ bit 112
  g1[4] = 0;                                    // tile_dim1/2 unused
  g1[5] = (int)n8;                              // tensor_dim0_stride lo
  g1[6] = 0;
  g1[7] = 0;
  i32x4 z4 = {0, 0, 0, 0};
  i32x8 z8 = {0, 0, 0, 0, 0, 0, 0, 0};
  __builtin_amdgcn_tensor_load_to_lds(g0, g1, z4, z4, z8, 0);
}

// ---------------------------------------------------------------------------
// Small prep kernels
// ---------------------------------------------------------------------------
__global__ void k_cam(const float* __restrict__ E_inv, const float* __restrict__ cam_w,
                      float* __restrict__ cE) {
  int bn = blockIdx.x, d = threadIdx.x;
  float s = 0.f;
#pragma unroll
  for (int i = 0; i < 4; ++i) s += E_inv[bn * 16 + i * 4 + 3] * cam_w[i * DIMC + d];
  cE[bn * DIMC + d] = s;
}

__global__ __launch_bounds__(128)
void k_img(const float* __restrict__ I_inv, const float* __restrict__ E_inv,
           const float* __restrict__ img_w, const float* __restrict__ cE,
           float* __restrict__ imgE) {
  long t = (long)blockIdx.x * 4 + (threadIdx.x >> 5);
  if (t >= TF) return;
  int lane = threadIdx.x & 31;
  int bn = (int)(t >> 8);
  int pix = (int)(t & 255);
  int h = pix >> 4, w = pix & 15;
  float gx = w * (256.0f / 15.0f);
  float gy = h * (256.0f / 15.0f);
  float cam[3];
#pragma unroll
  for (int i = 0; i < 3; ++i)
    cam[i] = I_inv[bn * 9 + i * 3 + 0] * gx + I_inv[bn * 9 + i * 3 + 1] * gy + I_inv[bn * 9 + i * 3 + 2];
  float dv[4];
#pragma unroll
  for (int i = 0; i < 4; ++i)
    dv[i] = E_inv[bn * 16 + i * 4 + 0] * cam[0] + E_inv[bn * 16 + i * 4 + 1] * cam[1] +
            E_inv[bn * 16 + i * 4 + 2] * cam[2] + E_inv[bn * 16 + i * 4 + 3];
  float e[4]; float s2 = 0.f;
#pragma unroll
  for (int j = 0; j < 4; ++j) {
    int c = lane * 4 + j;
    e[j] = dv[0] * img_w[c] + dv[1] * img_w[DIMC + c] + dv[2] * img_w[2 * DIMC + c] +
           dv[3] * img_w[3 * DIMC + c] - cE[bn * DIMC + c];
    s2 += e[j] * e[j];
  }
  float nrm = sqrtf(wsum32(s2)) + 1e-7f;
#pragma unroll
  for (int j = 0; j < 4; ++j) imgE[t * DIMC + lane * 4 + j] = e[j] / nrm;
}

__global__ void k_bev(const float* __restrict__ bev_grid, const float* __restrict__ bev_w,
                      const float* __restrict__ bev_b, float* __restrict__ wE) {
  long idx = (long)blockIdx.x * 256 + threadIdx.x;
  if (idx >= 4096L * DIMC) return;
  int c = (int)(idx & 127);
  long hw = idx >> 7;
  wE[idx] = bev_grid[hw] * bev_w[c] + bev_grid[4096 + hw] * bev_w[DIMC + c] + bev_b[c];
}

__global__ void k_xhw(const float* __restrict__ x, float* __restrict__ xHW) {
  long idx = (long)blockIdx.x * 256 + threadIdx.x;
  if (idx >= THW * DIMC) return;
  int c = (int)(idx & 127);
  long thw = idx >> 7;
  long hw = thw & 4095;
  int b = (int)(thw >> 12);
  xHW[idx] = x[((long)b * DIMC + c) * 4096 + hw];
}

__global__ void k_wt(const float* __restrict__ src, _Float16* __restrict__ dst, int K, int N) {
  long idx = (long)blockIdx.x * 256 + threadIdx.x;
  if (idx >= (long)K * N) return;
  int i = (int)(idx / N), j = (int)(idx % N);
  dst[(long)j * K + i] = (_Float16)src[idx];
}

// Vp[b,l][key][d] -> VpT[b,l][d][key]   (makes PV's B operand contiguous)
__global__ void k_vt(const _Float16* __restrict__ Vp, _Float16* __restrict__ VpT) {
  long idx = (long)blockIdx.x * 256 + threadIdx.x;
  if (idx >= TKV * DIMC) return;
  int d = (int)(idx & 127);
  long tk = idx >> 7;
  int key = (int)(tk % MK);
  long bl = tk / MK;
  VpT[(bl * DIMC + d) * MK + key] = Vp[idx];
}

// ---------------------------------------------------------------------------
// 3x3 SAME conv over 129 channels (feature + depth) + BN + ReLU
// ---------------------------------------------------------------------------
__global__ __launch_bounds__(128)
void k_conv(const float* __restrict__ feature, const float* __restrict__ depth,
            const float* __restrict__ dfw, const float* __restrict__ bg,
            const float* __restrict__ bb, const float* __restrict__ bm,
            const float* __restrict__ bv, float* __restrict__ feat2) {
  int blk = blockIdx.x;                 // bn*256 + h*16 + w
  int bn = blk >> 8, pix = blk & 255;
  int h = pix >> 4, w = pix & 15;
  __shared__ float patch[129 * 9];
  for (int li = threadIdx.x; li < 129 * 9; li += 128) {
    int i = li / 9, kk = li % 9;
    int hh = h + kk / 3 - 1, ww = w + kk % 3 - 1;
    float v = 0.f;
    if (hh >= 0 && hh < 16 && ww >= 0 && ww < 16)
      v = (i < 128) ? feature[((long)bn * 128 + i) * 256 + hh * 16 + ww]
                    : depth[(long)bn * 256 + hh * 16 + ww];
    patch[li] = v;
  }
  __syncthreads();
  int o = threadIdx.x;
  const float* wp = dfw + (long)o * 129 * 9;
  float acc = 0.f;
  for (int li = 0; li < 129 * 9; ++li) acc = fmaf(patch[li], wp[li], acc);
  float y = (acc - bm[o]) * (bg[o] * rsqrtf(bv[o] + EPSF)) + bb[o];
  feat2[(long)blk * 128 + o] = fmaxf(y, 0.f);
}

__global__ void k_act(const float* __restrict__ feat2,
                      const float* __restrict__ gP, const float* __restrict__ bP,
                      const float* __restrict__ mP, const float* __restrict__ vP,
                      const float* __restrict__ gL, const float* __restrict__ bL,
                      const float* __restrict__ mL, const float* __restrict__ vL,
                      _Float16* __restrict__ actP, _Float16* __restrict__ actL) {
  long idx = (long)blockIdx.x * 256 + threadIdx.x;
  if (idx >= (long)TF * DIMC) return;
  int c = (int)(idx & 127);
  float f = feat2[idx];
  float p = fmaxf((f - mP[c]) * (gP[c] * rsqrtf(vP[c] + EPSF)) + bP[c], 0.f);
  float l = fmaxf((f - mL[c]) * (gL[c] * rsqrtf(vL[c] + EPSF)) + bL[c], 0.f);
  actP[idx] = (_Float16)p;
  actL[idx] = (_Float16)l;
}

// ---------------------------------------------------------------------------
// Generic WMMA GEMM: out[T][N] = A[T][K](f16) * W(col-major f16 [N][K])
// One wave -> 16x16 tile; block of 4 waves -> 64 rows sharing one B tile.
// The 16xK B tile (contiguous in Wt) is DMA'd to LDS by the TDM once per
// block; all 4 waves read it back as ds loads. T must be a multiple of 64.
// ---------------------------------------------------------------------------
template <bool HAS_BIAS, bool HAS_ADD, int ACT, bool OUT_F16>
__global__ __launch_bounds__(128)
void k_gemm(const _Float16* __restrict__ A, const _Float16* __restrict__ Wt,
            const float* __restrict__ bias, const float* __restrict__ add,
            void* __restrict__ outp, long T, int K, int N) {
  __shared__ alignas(256) _Float16 Bs[16 * 256];  // only LDS object -> offset 0
  int wave = threadIdx.x >> 5;
  int lane = threadIdx.x & 31;
  long t0 = ((long)blockIdx.x * 4 + wave) * 16;
  int n0 = blockIdx.y * 16;
  if (wave == 0) {
    tdm_load_tile(Wt + (long)n0 * K, 0u, (unsigned)(16 * K * 2 / 8));
    __builtin_amdgcn_s_wait_tensorcnt(0);
  }
  __syncthreads();
  v8f acc = {};
  for (int kk = 0; kk < K; kk += 32) {
    __builtin_prefetch(A + t0 * K + kk + 128, 0, 0);   // global_prefetch_b8
    v16h a = wmma_load_a16(A + t0 * K + kk, K);
    v16h b = wmma_load_b_cm(Bs + kk, K);
    acc = __builtin_amdgcn_wmma_f32_16x16x32_f16(false, a, false, b, (short)0, acc, false, false);
  }
  int n = n0 + (lane & 15);
  int hf2 = lane >> 4;
  float bvv = HAS_BIAS ? bias[n] : 0.f;
#pragma unroll
  for (int v = 0; v < 8; ++v) {
    long t = t0 + hf2 * 8 + v;
    float val = acc[v] + bvv;
    if (HAS_ADD) val += add[t * N + n];
    if (ACT == 1) val = gelu_exact(val);
    if (OUT_F16) ((_Float16*)outp)[t * N + n] = (_Float16)val;
    else         ((float*)outp)[t * N + n] = val;
  }
}

// ---------------------------------------------------------------------------
// LayerNorm / token-build kernels (one wave per token, 4 channels per lane)
// ---------------------------------------------------------------------------
__device__ inline void ln_write(const float v[4], const float* g, const float* bt,
                                _Float16* dst, int lane) {
  float s = v[0] + v[1] + v[2] + v[3];
  float mean = wsum32(s) * (1.f / 128.f);
  float q = 0.f;
#pragma unroll
  for (int j = 0; j < 4; ++j) { float d = v[j] - mean; q += d * d; }
  float inv = rsqrtf(wsum32(q) * (1.f / 128.f) + EPSF);
#pragma unroll
  for (int j = 0; j < 4; ++j) {
    int c = lane * 4 + j;
    dst[c] = (_Float16)((v[j] - mean) * inv * g[c] + bt[c]);
  }
}

__global__ __launch_bounds__(128)
void k_ln_f16(const float* __restrict__ in, const float* __restrict__ g,
              const float* __restrict__ bt, _Float16* __restrict__ out, long T) {
  long t = (long)blockIdx.x * 4 + (threadIdx.x >> 5);
  if (t >= T) return;
  int lane = threadIdx.x & 31;
  float v[4];
#pragma unroll
  for (int j = 0; j < 4; ++j) v[j] = in[t * DIMC + lane * 4 + j];
  ln_write(v, g, bt, out + t * DIMC, lane);
}

// pass-1 queries: (bev_e normalized) + x, then LN(a1_q)
__global__ __launch_bounds__(128)
void k_build_q1(const float* __restrict__ wE, const float* __restrict__ cE,
                const float* __restrict__ xHW, const float* __restrict__ g,
                const float* __restrict__ bt, _Float16* __restrict__ qtok) {
  long t = (long)blockIdx.x * 4 + (threadIdx.x >> 5);
  if (t >= TQ) return;
  int lane = threadIdx.x & 31;
  int m = (int)(t % MQ); long rest = t / MQ;
  int l = (int)(rest & 15); int b = (int)(rest >> 4);
  int n = m >> 8, r = m & 255;
  int w1 = r >> 4, w2 = r & 15;
  int xw = l >> 2, yw = l & 3;
  int h = xw * 16 + w1, w = yw * 16 + w2;
  int hw = h * 64 + w;
  int bn = b * NCAM + n;
  float e[4]; float s2 = 0.f;
#pragma unroll
  for (int j = 0; j < 4; ++j) {
    int c = lane * 4 + j;
    e[j] = wE[(long)hw * DIMC + c] - cE[bn * DIMC + c];
    s2 += e[j] * e[j];
  }
  float nrm = sqrtf(wsum32(s2)) + 1e-7f;
  float v[4];
#pragma unroll
  for (int j = 0; j < 4; ++j) {
    int c = lane * 4 + j;
    v[j] = e[j] / nrm + xHW[((long)b * 4096 + hw) * DIMC + c];
  }
  ln_write(v, g, bt, qtok + t * DIMC, lane);
}

// pass-2 queries: broadcast z over cameras, LN(a2_q)
__global__ __launch_bounds__(128)
void k_build_q2(const float* __restrict__ zHW, const float* __restrict__ g,
                const float* __restrict__ bt, _Float16* __restrict__ qtok) {
  long t = (long)blockIdx.x * 4 + (threadIdx.x >> 5);
  if (t >= TQ) return;
  int lane = threadIdx.x & 31;
  int m = (int)(t % MQ); long rest = t / MQ;
  int l = (int)(rest & 15); int b = (int)(rest >> 4);
  int r = m & 255;
  int w1 = r >> 4, w2 = r & 15;
  int xw = l >> 2, yw = l & 3;
  int hw = (xw * 16 + w1) * 64 + (yw * 16 + w2);
  const float* p = zHW + ((long)b * 4096 + hw) * DIMC;
  float v[4];
#pragma unroll
  for (int j = 0; j < 4; ++j) v[j] = p[lane * 4 + j];
  ln_write(v, g, bt, qtok + t * DIMC, lane);
}

// merged+LN'ed K/V tokens; pass selects window mapping (1=local, 2=swapped)
__global__ __launch_bounds__(128)
void k_build_kv(const float* __restrict__ src, const float* __restrict__ g,
                const float* __restrict__ bt, _Float16* __restrict__ dst, int pass) {
  long t = (long)blockIdx.x * 4 + (threadIdx.x >> 5);
  if (t >= TKV) return;
  int lane = threadIdx.x & 31;
  int m = (int)(t % MK); long rest = t / MK;
  int l = (int)(rest & 15); int b = (int)(rest >> 4);
  int n = m >> 4, r = m & 15;
  int h, w;
  if (pass == 1) { int xw = l >> 2, yw = l & 3, dh = r >> 2, dw = r & 3; h = xw * 4 + dh; w = yw * 4 + dw; }
  else           { int dh = l >> 2, dw = l & 3, xw = r >> 2, yw = r & 3; h = xw * 4 + dh; w = yw * 4 + dw; }
  const float* p = src + ((long)(b * NCAM + n) * 256 + h * 16 + w) * DIMC;
  float v[4];
#pragma unroll
  for (int j = 0; j < 4; ++j) v[j] = p[lane * 4 + j];
  ln_write(v, g, bt, dst + t * DIMC, lane);
}

// ---------------------------------------------------------------------------
// Windowed multi-head attention: one wave per (b, l, head, 16-query tile)
// ---------------------------------------------------------------------------
__global__ __launch_bounds__(32)
void k_attn(const _Float16* __restrict__ Qp, const _Float16* __restrict__ Kp,
            const _Float16* __restrict__ VpT, _Float16* __restrict__ aOut) {
  int id = blockIdx.x;
  int qt = id % (MQ / 16); id /= (MQ / 16);
  int hd = id & 3; id >>= 2;
  int l = id & 15; id >>= 4;
  int b = id;
  int lane = threadIdx.x;
  long bl = (long)(b * 16 + l);
  const _Float16* Qb = Qp + (bl * MQ + qt * 16) * DIMC + hd * DHEAD;
  const _Float16* Kb = Kp + bl * MK * DIMC + hd * DHEAD;
  const _Float16* Vt = VpT + (bl * DIMC + hd * DHEAD) * MK;   // [d][key]
  __shared__ alignas(32) float    sc[16 * MK];
  __shared__ alignas(32) _Float16 pr[16 * MK];

  v16h a = wmma_load_a16(Qb, DIMC);          // 16x32 query tile
  int n = lane & 15, hf2 = lane >> 4;
#pragma unroll
  for (int nt = 0; nt < MK / 16; ++nt) {     // 6 key tiles: scores = Q * K^T
    v16h bm = wmma_load_b_cm(Kb + (long)nt * 16 * DIMC, DIMC);
    v8f acc = {};
    acc = __builtin_amdgcn_wmma_f32_16x16x32_f16(false, a, false, bm, (short)0, acc, false, false);
#pragma unroll
    for (int v = 0; v < 8; ++v) sc[(hf2 * 8 + v) * MK + nt * 16 + n] = acc[v];
  }
  __syncthreads();
  const float scale = 0.17677669529663687f;   // 32^-0.5
  if (lane < 16) {
    int r = lane;
    float mx = -1e30f;
    for (int k = 0; k < MK; ++k) mx = fmaxf(mx, sc[r * MK + k]);
    float s = 0.f;
    for (int k = 0; k < MK; ++k) { float e = expf((sc[r * MK + k] - mx) * scale); sc[r * MK + k] = e; s += e; }
    float inv = 1.f / s;
    for (int k = 0; k < MK; ++k) pr[r * MK + k] = (_Float16)(sc[r * MK + k] * inv);
  }
  __syncthreads();
#pragma unroll
  for (int nt = 0; nt < 2; ++nt) {           // out = P * V   (16 x 32)
    v8f acc = {};
#pragma unroll
    for (int kc = 0; kc < 3; ++kc) {
      v16h pa = wmma_load_a16(pr + kc * 32, MK);
      v16h vb = wmma_load_b_cm(Vt + (long)nt * 16 * MK + kc * 32, MK);
      acc = __builtin_amdgcn_wmma_f32_16x16x32_f16(false, pa, false, vb, (short)0, acc, false, false);
    }
    _Float16* Ob = aOut + (bl * MQ + qt * 16) * DIMC + hd * DHEAD + nt * 16;
#pragma unroll
    for (int v = 0; v < 8; ++v) Ob[(long)(hf2 * 8 + v) * DIMC + n] = (_Float16)acc[v];
  }
}

// mean over cameras, scatter merged tokens -> spatial (b,h,w) token order
__global__ void k_rmean(const _Float16* __restrict__ aOut, _Float16* __restrict__ aMean) {
  long idx = (long)blockIdx.x * 256 + threadIdx.x;
  if (idx >= THW * DIMC) return;
  int c = (int)(idx & 127);
  long thw = idx >> 7;
  int w = (int)(thw & 63); long r2 = thw >> 6;
  int h = (int)(r2 & 63); int b = (int)(r2 >> 6);
  int xw = h >> 4, w1 = h & 15, yw = w >> 4, w2 = w & 15;
  int l = xw * 4 + yw;
  float s = 0.f;
#pragma unroll
  for (int n = 0; n < NCAM; ++n)
    s += (float)aOut[(((long)(b * 16 + l)) * MQ + n * 256 + w1 * 16 + w2) * DIMC + c];
  aMean[idx] = (_Float16)(s * (1.f / 6.f));
}

// final LN + transpose to (b, c, h, w)
__global__ __launch_bounds__(128)
void k_ln_out(const float* __restrict__ in, const float* __restrict__ g,
              const float* __restrict__ bt, float* __restrict__ out) {
  long t = (long)blockIdx.x * 4 + (threadIdx.x >> 5);
  if (t >= THW) return;
  int lane = threadIdx.x & 31;
  const float* p = in + t * DIMC;
  float v[4];
#pragma unroll
  for (int j = 0; j < 4; ++j) v[j] = p[lane * 4 + j];
  float s = v[0] + v[1] + v[2] + v[3];
  float mean = wsum32(s) * (1.f / 128.f);
  float q = 0.f;
#pragma unroll
  for (int j = 0; j < 4; ++j) { float d = v[j] - mean; q += d * d; }
  float inv = rsqrtf(wsum32(q) * (1.f / 128.f) + EPSF);
  long hw = t & 4095; int b = (int)(t >> 12);
#pragma unroll
  for (int j = 0; j < 4; ++j) {
    int c = lane * 4 + j;
    out[((long)b * DIMC + c) * 4096 + hw] = (v[j] - mean) * inv * g[c] + bt[c];
  }
}

// ---------------------------------------------------------------------------
// Host side
// ---------------------------------------------------------------------------
#define EW(total) dim3((unsigned)(((long)(total) + 255) / 256)), dim3(256)

extern "C" void kernel_launch(void* const* d_in, const int* in_sizes, int n_in,
                              void* d_out, int out_size, void* d_ws, size_t ws_size,
                              hipStream_t stream) {
  // ---- inputs (setup_inputs dict order; params flattened in insertion order)
  const float* x        = (const float*)d_in[0];
  const float* feature  = (const float*)d_in[1];
  const float* I_inv    = (const float*)d_in[2];
  const float* E_inv    = (const float*)d_in[3];
  const float* bev_grid = (const float*)d_in[4];
  const float* depth    = (const float*)d_in[5];
  const float* cam_w = (const float*)d_in[6], *img_w = (const float*)d_in[7];
  const float* bev_w = (const float*)d_in[8], *bev_b = (const float*)d_in[9];
  const float* flin_w = (const float*)d_in[10], *fproj_w = (const float*)d_in[11];
  const float* df_w = (const float*)d_in[12];
  const float* flin_g = (const float*)d_in[13], *flin_b = (const float*)d_in[14];
  const float* flin_m = (const float*)d_in[15], *flin_v = (const float*)d_in[16];
  const float* fpro_g = (const float*)d_in[17], *fpro_b = (const float*)d_in[18];
  const float* fpro_m = (const float*)d_in[19], *fpro_v = (const float*)d_in[20];
  const float* dfbn_g = (const float*)d_in[21], *dfbn_b = (const float*)d_in[22];
  const float* dfbn_m = (const float*)d_in[23], *dfbn_v = (const float*)d_in[24];
  const float* a1q_g = (const float*)d_in[25], *a1q_b = (const float*)d_in[26];
  const float* a1q_w = (const float*)d_in[27], *a1q_bias = (const float*)d_in[28];
  const float* a1k_g = (const float*)d_in[29], *a1k_b = (const float*)d_in[30];
  const float* a1k_w = (const float*)d_in[31], *a1k_bias = (const float*)d_in[32];
  const float* a1v_g = (const float*)d_in[33], *a1v_b = (const float*)d_in[34];
  const float* a1v_w = (const float*)d_in[35], *a1v_bias = (const float*)d_in[36];
  const float* a1_pw = (const float*)d_in[37], *a1_pb = (const float*)d_in[38];
  const float* a2q_g = (const float*)d_in[39], *a2q_b = (const float*)d_in[40];
  const float* a2q_w = (const float*)d_in[41], *a2q_bias = (const float*)d_in[42];
  const float* a2k_g = (const float*)d_in[43], *a2k_b = (const float*)d_in[44];
  const float* a2k_w = (const float*)d_in[45], *a2k_bias = (const float*)d_in[46];
  const float* a2v_g = (const float*)d_in[47], *a2v_b = (const float*)d_in[48];
  const float* a2v_w = (const float*)d_in[49], *a2v_bias = (const float*)d_in[50];
  const float* a2_pw = (const float*)d_in[51], *a2_pb = (const float*)d_in[52];
  const float* pre1_g = (const float*)d_in[53], *pre1_b = (const float*)d_in[54];
  const float* pre2_g = (const float*)d_in[55], *pre2_b = (const float*)d_in[56];
  const float* post_g = (const float*)d_in[57], *post_b = (const float*)d_in[58];
  const float* m1_w1 = (const float*)d_in[59], *m1_b1 = (const float*)d_in[60];
  const float* m1_w2 = (const float*)d_in[61], *m1_b2 = (const float*)d_in[62];
  const float* m2_w1 = (const float*)d_in[63], *m2_b1 = (const float*)d_in[64];
  const float* m2_w2 = (const float*)d_in[65], *m2_b2 = (const float*)d_in[66];
  float* out = (float*)d_out;

  // ---- workspace carve-up
  char* wsb = (char*)d_ws;
  size_t off = 0;
#define WS_ALLOC(TYPE, NAME, CNT) \
  TYPE* NAME = (TYPE*)(wsb + off); off = (off + (size_t)(CNT) * sizeof(TYPE) + 255) & ~(size_t)255;
  WS_ALLOC(float, cE, 24 * DIMC)
  WS_ALLOC(float, wE, 4096 * DIMC)
  WS_ALLOC(float, imgE, (size_t)TF * DIMC)
  WS_ALLOC(float, xHW, (size_t)THW * DIMC)
  WS_ALLOC(float, feat2, (size_t)TF * DIMC)
  WS_ALLOC(_Float16, actP, (size_t)TF * DIMC)
  WS_ALLOC(_Float16, actL, (size_t)TF * DIMC)
  WS_ALLOC(float, Ktok, (size_t)TF * DIMC)
  WS_ALLOC(float, Vtok, (size_t)TF * DIMC)
  WS_ALLOC(_Float16, WtQ1, DIMC * DIMC) WS_ALLOC(_Float16, WtK1, DIMC * DIMC)
  WS_ALLOC(_Float16, WtV1, DIMC * DIMC) WS_ALLOC(_Float16, WtP1, DIMC * DIMC)
  WS_ALLOC(_Float16, WtQ2, DIMC * DIMC) WS_ALLOC(_Float16, WtK2, DIMC * DIMC)
  WS_ALLOC(_Float16, WtV2, DIMC * DIMC) WS_ALLOC(_Float16, WtP2, DIMC * DIMC)
  WS_ALLOC(_Float16, WtM11, DIMC * 256) WS_ALLOC(_Float16, WtM12, 256 * DIMC)
  WS_ALLOC(_Float16, WtM21, DIMC * 256) WS_ALLOC(_Float16, WtM22, 256 * DIMC)
  WS_ALLOC(_Float16, WtFP, DIMC * DIMC) WS_ALLOC(_Float16, WtFL, DIMC * DIMC)
  WS_ALLOC(_Float16, qtok, (size_t)TQ * DIMC)
  WS_ALLOC(_Float16, Qp, (size_t)TQ * DIMC)
  WS_ALLOC(_Float16, Km, (size_t)TKV * DIMC)
  WS_ALLOC(_Float16, Vm, (size_t)TKV * DIMC)
  WS_ALLOC(_Float16, Kp, (size_t)TKV * DIMC)
  WS_ALLOC(_Float16, Vp, (size_t)TKV * DIMC)
  WS_ALLOC(_Float16, VpT, (size_t)TKV * DIMC)
  WS_ALLOC(_Float16, aOut, (size_t)TQ * DIMC)
  WS_ALLOC(_Float16, aMean, (size_t)THW * DIMC)
  WS_ALLOC(float, zA, (size_t)THW * DIMC)
  WS_ALLOC(float, zB, (size_t)THW * DIMC)
  WS_ALLOC(_Float16, lnbuf, (size_t)THW * DIMC)
  WS_ALLOC(_Float16, hbuf, (size_t)THW * 256)
#undef WS_ALLOC
  (void)in_sizes; (void)n_in; (void)out_size; (void)ws_size;

  // ---- prep
  k_cam<<<24, 128, 0, stream>>>(E_inv, cam_w, cE);
  k_img<<<TF / 4, 128, 0, stream>>>(I_inv, E_inv, img_w, cE, imgE);
  k_bev<<<EW(4096L * DIMC), 0, stream>>>(bev_grid, bev_w, bev_b, wE);
  k_xhw<<<EW(THW * DIMC), 0, stream>>>(x, xHW);
  k_conv<<<TF, 128, 0, stream>>>(feature, depth, df_w, dfbn_g, dfbn_b, dfbn_m, dfbn_v, feat2);
  k_act<<<EW((long)TF * DIMC), 0, stream>>>(feat2, fpro_g, fpro_b, fpro_m, fpro_v,
                                            flin_g, flin_b, flin_m, flin_v, actP, actL);
  // weight transposes -> f16 col-major
  k_wt<<<EW(128 * 128), 0, stream>>>(a1q_w, WtQ1, 128, 128);
  k_wt<<<EW(128 * 128), 0, stream>>>(a1k_w, WtK1, 128, 128);
  k_wt<<<EW(128 * 128), 0, stream>>>(a1v_w, WtV1, 128, 128);
  k_wt<<<EW(128 * 128), 0, stream>>>(a1_pw, WtP1, 128, 128);
  k_wt<<<EW(128 * 128), 0, stream>>>(a2q_w, WtQ2, 128, 128);
  k_wt<<<EW(128 * 128), 0, stream>>>(a2k_w, WtK2, 128, 128);
  k_wt<<<EW(128 * 128), 0, stream>>>(a2v_w, WtV2, 128, 128);
  k_wt<<<EW(128 * 128), 0, stream>>>(a2_pw, WtP2, 128, 128);
  k_wt<<<EW(128 * 256), 0, stream>>>(m1_w1, WtM11, 128, 256);
  k_wt<<<EW(256 * 128), 0, stream>>>(m1_w2, WtM12, 256, 128);
  k_wt<<<EW(128 * 256), 0, stream>>>(m2_w1, WtM21, 128, 256);
  k_wt<<<EW(256 * 128), 0, stream>>>(m2_w2, WtM22, 256, 128);
  k_wt<<<EW(128 * 128), 0, stream>>>(fproj_w, WtFP, 128, 128);
  k_wt<<<EW(128 * 128), 0, stream>>>(flin_w, WtFL, 128, 128);

  // key/value features: key = imgE + act_fproj @ W ; val = act_flin @ W
  k_gemm<false, true, 0, false><<<dim3(TF / 64, 8), 128, 0, stream>>>(actP, WtFP, nullptr, imgE, Ktok, TF, 128, 128);
  k_gemm<false, false, 0, false><<<dim3(TF / 64, 8), 128, 0, stream>>>(actL, WtFL, nullptr, nullptr, Vtok, TF, 128, 128);

  for (int pass = 1; pass <= 2; ++pass) {
    const _Float16 *WQ = pass == 1 ? WtQ1 : WtQ2, *WK = pass == 1 ? WtK1 : WtK2;
    const _Float16 *WV = pass == 1 ? WtV1 : WtV2, *WP = pass == 1 ? WtP1 : WtP2;
    const float *qbias = pass == 1 ? a1q_bias : a2q_bias;
    const float *kbias = pass == 1 ? a1k_bias : a2k_bias;
    const float *vbias = pass == 1 ? a1v_bias : a2v_bias;
    const float *pbias = pass == 1 ? a1_pb : a2_pb;
    const float *skip  = pass == 1 ? xHW : zB;
    if (pass == 1)
      k_build_q1<<<TQ / 4, 128, 0, stream>>>(wE, cE, xHW, a1q_g, a1q_b, qtok);
    else
      k_build_q2<<<TQ / 4, 128, 0, stream>>>(zB, a2q_g, a2q_b, qtok);
    k_build_kv<<<TKV / 4, 128, 0, stream>>>(Ktok, pass == 1 ? a1k_g : a2k_g,
                                            pass == 1 ? a1k_b : a2k_b, Km, pass);
    k_build_kv<<<TKV / 4, 128, 0, stream>>>(Vtok, pass == 1 ? a1v_g : a2v_g,
                                            pass == 1 ? a1v_b : a2v_b, Vm, pass);
    // projections (f16 out for attention)
    k_gemm<true, false, 0, true><<<dim3(TQ / 64, 8), 128, 0, stream>>>(qtok, WQ, qbias, nullptr, Qp, TQ, 128, 128);
    k_gemm<true, false, 0, true><<<dim3(TKV / 64, 8), 128, 0, stream>>>(Km, WK, kbias, nullptr, Kp, TKV, 128, 128);
    k_gemm<true, false, 0, true><<<dim3(TKV / 64, 8), 128, 0, stream>>>(Vm, WV, vbias, nullptr, Vp, TKV, 128, 128);
    k_vt<<<EW(TKV * DIMC), 0, stream>>>(Vp, VpT);
    // attention
    k_attn<<<BB * LWIN * NHEAD * (MQ / 16), 32, 0, stream>>>(Qp, Kp, VpT, aOut);
    // camera mean (commutes with linear pw), then pw projection + skip
    k_rmean<<<EW(THW * DIMC), 0, stream>>>(aOut, aMean);
    k_gemm<true, true, 0, false><<<dim3(THW / 64, 8), 128, 0, stream>>>(aMean, WP, pbias, skip, zA, THW, 128, 128);
    // MLP block: z = z + W2(gelu(W1(LN(z))))
    k_ln_f16<<<THW / 4, 128, 0, stream>>>(zA, pass == 1 ? pre1_g : pre2_g,
                                          pass == 1 ? pre1_b : pre2_b, lnbuf, THW);
    k_gemm<true, false, 1, true><<<dim3(THW / 64, 16), 128, 0, stream>>>(
        lnbuf, pass == 1 ? WtM11 : WtM21, pass == 1 ? m1_b1 : m2_b1, nullptr, hbuf, THW, 128, 256);
    k_gemm<true, true, 0, false><<<dim3(THW / 64, 8), 128, 0, stream>>>(
        hbuf, pass == 1 ? WtM12 : WtM22, pass == 1 ? m1_b2 : m2_b2, zA, zB, THW, 256, 128);
  }
  // final post-LN, transpose to (b, c, h, w)
  k_ln_out<<<THW / 4, 128, 0, stream>>>(zB, post_g, post_b, out);
}
